// LSTM_7112465842330
// MI455X (gfx1250) — matrix operand order repaired and checked by reading, not compiled, not run on hardware
//
#include <hip/hip_runtime.h>
#include <math.h>

// MI455X / gfx1250, wave32. bf16 WMMA 16x16x32, f32 accumulate.

typedef __attribute__((ext_vector_type(16))) __bf16 v16bf;
typedef __attribute__((ext_vector_type(8)))  __bf16 v8bf;
typedef __attribute__((ext_vector_type(8)))  float  v8f;

#define B_  32
#define T_  512
#define D_  512
#define H_  512
#define G4_ 2048   // 4*H
#define L_  2

// ---------------- fragment loaders (ISA 7.12.2 layouts) ----------------
// A operand (16x32 bf16): per lane, 8 contiguous bf16 at k-offset kb and 8 at kb+16,
// where kb = (lane>=16 ? 8 : 0); row = lane & 15.
__device__ __forceinline__ v16bf ldA(const __bf16* p) {
  v8bf lo = *reinterpret_cast<const v8bf*>(p);
  v8bf hi = *reinterpret_cast<const v8bf*>(p + 16);
  v16bf f;
#pragma unroll
  for (int i = 0; i < 8; ++i) { f[i] = lo[i]; f[i + 8] = hi[i]; }
  return f;
}
// B operand (32x16 bf16): per lane (= column n), 8 contiguous bf16 at k-offset kb
// and 8 at kb+16, kb = (lane>=16 ? 8 : 0). Source is W stored [N,K] row-major.
__device__ __forceinline__ v16bf ldB(const __bf16* p) {
  v8bf lo = *reinterpret_cast<const v8bf*>(p);
  v8bf hi = *reinterpret_cast<const v8bf*>(p + 16);
  v16bf f;
#pragma unroll
  for (int i = 0; i < 8; ++i) { f[i] = lo[i]; f[i + 8] = hi[i]; }
  return f;
}

template <int KTILES>
__device__ __forceinline__ v8f gemm_part(const __bf16* aBase, const __bf16* bBase, v8f acc) {
#pragma unroll 4
  for (int kt = 0; kt < KTILES; ++kt) {
    v16bf a = ldA(aBase + kt * 32);
    v16bf b = ldB(bBase + kt * 32);
    acc = __builtin_amdgcn_wmma_f32_16x16x32_bf16(false, a, false, b, (short)0, acc,
                                                  false, false);
  }
  return acc;
}

// ---------------- prep: f32->bf16 weights, seed h0/c0 ----------------
__global__ void lstm_prep(const float* __restrict__ Wx, const float* __restrict__ Wh,
                          const float* __restrict__ h0, const float* __restrict__ c0,
                          __bf16* __restrict__ WxBf, __bf16* __restrict__ WhBf,
                          __bf16* __restrict__ hseq,  // [L][T+1][B][H] bf16
                          float* __restrict__ cst)    // [L][B][H] f32
{
  size_t i = (size_t)blockIdx.x * blockDim.x + threadIdx.x;
  const size_t nW = (size_t)L_ * G4_ * D_;
  if (i < nW) { WxBf[i] = (__bf16)Wx[i]; WhBf[i] = (__bf16)Wh[i]; }
  if (i < (size_t)L_ * B_ * H_) {
    int j = (int)(i & (H_ - 1));
    int l = (int)((i >> 9) & 1);   // h0/c0 layout is [B, L, H]
    int b = (int)(i >> 10);
    float hv = h0[((size_t)b * L_ + l) * H_ + j];
    float cv = c0[((size_t)b * L_ + l) * H_ + j];
    hseq[(size_t)l * (T_ + 1) * B_ * H_ + (size_t)b * H_ + j] = (__bf16)hv; // slot 0
    cst[(size_t)l * B_ * H_ + (size_t)b * H_ + j] = cv;
  }
}

// ---------------- one LSTM timestep for one layer ----------------
// grid = (8, 2): blockIdx.x = 64-wide hidden-column slab, blockIdx.y = 16-row batch tile
// block = 512 threads = 16 waves; wave w -> gate (w>>2), 16-col N-tile (w&3) in slab.
__global__ __launch_bounds__(512, 1)
void lstm_step(int t, int layer, int is_last,
               const float*  __restrict__ x,      // [B,T,D] f32 (layer-0 input)
               const __bf16* __restrict__ Wxl,    // [4H,D] bf16
               const __bf16* __restrict__ Whl,    // [4H,H] bf16
               const float*  __restrict__ bxl,    // [4H]
               const float*  __restrict__ bhl,    // [4H]
               const __bf16* __restrict__ hin,    // layer-0 h history (layer-1 input)
               __bf16*       __restrict__ hseq_l, // this layer's h history [T+1,B,H]
               float*        __restrict__ cst_l,  // [B,H]
               float*        __restrict__ out,    // [B,T,H] (layer 1 writes)
               float*        __restrict__ hn,     // [L,B,H]
               float*        __restrict__ cn)     // [L,B,H]
{
  __shared__ __bf16 xbuf[16 * 520];   // x_t tile, padded stride (bank spread)
  __shared__ float  gbuf[16 * 260];   // gate pre-activations for this block

  const int tid  = threadIdx.x;
  const int lane = tid & 31;
  const int w    = tid >> 5;
  const int nblk = blockIdx.x;        // 0..7
  const int b0   = blockIdx.y * 16;

  // Stage + convert x_t rows (layer 0 only): 16 rows x 512 f32 -> bf16 LDS.
  if (layer == 0) {
#pragma unroll
    for (int i = 0; i < 16; ++i) {
      int e = tid + i * 512;          // 0..8191
      int r = e >> 9, d = e & 511;
      xbuf[r * 520 + d] = (__bf16)x[(((size_t)(b0 + r)) * T_ + t) * D_ + d];
    }
  }
  __syncthreads();

  const int rA   = lane & 15;
  const int kbA  = (lane >> 4) * 8;   // A k-offset split
  const int kbB  = (lane >> 4) * 8;   // B k-offset split
  const int gate = w >> 2;
  const int sub  = w & 3;
  const int wrow = gate * H_ + nblk * 64 + sub * 16 + (lane & 15); // W row = out col

  v8f acc = {};

  // x-part: K = 512
  if (layer == 0) {
    acc = gemm_part<D_ / 32>(&xbuf[rA * 520 + kbA],
                             Wxl + (size_t)wrow * D_ + kbB, acc);
  } else {
    // layer-1 input = layer-0 h at time t = hin slot (t+1), already bf16 A-layout
    acc = gemm_part<D_ / 32>(hin + ((size_t)(t + 1) * B_ + b0 + rA) * H_ + kbA,
                             Wxl + (size_t)wrow * D_ + kbB, acc);
  }
  // h-part: K = 512, h_{t-1} = own history slot t
  acc = gemm_part<H_ / 32>(hseq_l + ((size_t)t * B_ + b0 + rA) * H_ + kbA,
                           Whl + (size_t)wrow * H_ + kbB, acc);

  // D layout: vgpr v, lanes 0-15 -> M=v, N=lane; lanes 16-31 -> M=v+8, N=lane-16
  {
    const int mrow = (lane >> 4) * 8;
    const int col  = w * 16 + (lane & 15);
#pragma unroll
    for (int v = 0; v < 8; ++v) gbuf[(mrow + v) * 260 + col] = acc[v];
  }
  __syncthreads();

  // Pointwise LSTM cell: 16 rows x 64 hidden cols = 1024 elems, 2 per thread.
#pragma unroll
  for (int i = 0; i < 2; ++i) {
    int e  = tid + i * 512;
    int r  = e >> 6;                  // 0..15
    int jl = e & 63;
    int j  = nblk * 64 + jl;          // hidden index
    int b  = b0 + r;
    float gi = gbuf[r * 260 + jl]        + bxl[0 * H_ + j] + bhl[0 * H_ + j];
    float gf = gbuf[r * 260 + 64 + jl]   + bxl[1 * H_ + j] + bhl[1 * H_ + j];
    float gg = gbuf[r * 260 + 128 + jl]  + bxl[2 * H_ + j] + bhl[2 * H_ + j];
    float go = gbuf[r * 260 + 192 + jl]  + bxl[3 * H_ + j] + bhl[3 * H_ + j];
    float si = 1.0f / (1.0f + __expf(-gi));
    float sf = 1.0f / (1.0f + __expf(-gf));
    float so = 1.0f / (1.0f + __expf(-go));
    float tg = tanhf(gg);
    float c_old = cst_l[(size_t)b * H_ + j];
    float c_new = sf * c_old + si * tg;
    float h_new = so * tanhf(c_new);
    cst_l[(size_t)b * H_ + j] = c_new;
    hseq_l[((size_t)(t + 1) * B_ + b) * H_ + j] = (__bf16)h_new;
    if (layer == 1) out[(((size_t)b) * T_ + t) * H_ + j] = h_new;
    if (is_last) {
      hn[((size_t)layer * B_ + b) * H_ + j] = h_new;
      cn[((size_t)layer * B_ + b) * H_ + j] = c_new;
    }
  }
}

// ---------------- host ----------------
extern "C" void kernel_launch(void* const* d_in, const int* in_sizes, int n_in,
                              void* d_out, int out_size, void* d_ws, size_t ws_size,
                              hipStream_t stream) {
  (void)in_sizes; (void)n_in; (void)out_size;
  const float* x  = (const float*)d_in[0];
  const float* h0 = (const float*)d_in[1];
  const float* c0 = (const float*)d_in[2];
  const float* Wx = (const float*)d_in[3];
  const float* bx = (const float*)d_in[4];
  const float* Wh = (const float*)d_in[5];
  const float* bh = (const float*)d_in[6];
  float* out = (float*)d_out;

  const size_t WSZ    = (size_t)G4_ * D_;        // per-layer weight elems
  const size_t HS     = (size_t)B_ * H_;         // one h/c slot
  const size_t HSEQ_L = (size_t)(T_ + 1) * HS;   // per-layer h history

  // Workspace layout (needs ~42.1 MB):
  char* p = (char*)d_ws;
  __bf16* WxBf = (__bf16*)p;  p += (size_t)L_ * WSZ * sizeof(__bf16);
  __bf16* WhBf = (__bf16*)p;  p += (size_t)L_ * WSZ * sizeof(__bf16);
  __bf16* hseq = (__bf16*)p;  p += (size_t)L_ * HSEQ_L * sizeof(__bf16);
  float*  cst  = (float*)p;
  (void)ws_size;

  const int nW = L_ * G4_ * D_;   // 2,097,152
  lstm_prep<<<dim3((nW + 255) / 256), dim3(256), 0, stream>>>(
      Wx, Wh, h0, c0, WxBf, WhBf, hseq, cst);

  float* hn = out + (size_t)B_ * T_ * H_;
  float* cn = hn + (size_t)L_ * B_ * H_;

  for (int l = 0; l < L_; ++l) {
    const __bf16* Wxl = WxBf + (size_t)l * WSZ;
    const __bf16* Whl = WhBf + (size_t)l * WSZ;
    const float*  bxl = bx + (size_t)l * G4_;
    const float*  bhl = bh + (size_t)l * G4_;
    __bf16* hseq_l = hseq + (size_t)l * HSEQ_L;
    const __bf16* hin = hseq;   // layer-0 history feeds layer 1
    float* cst_l = cst + (size_t)l * HS;
    for (int t = 0; t < T_; ++t) {
      lstm_step<<<dim3(8, 2), dim3(512), 0, stream>>>(
          t, l, (t == T_ - 1) ? 1 : 0, x, Wxl, Whl, bxl, bhl, hin,
          hseq_l, cst_l, out, hn, cn);
    }
  }
}
